// TrigramMoE_34737695490678
// MI455X (gfx1250) — compile-verified, matrix operand order checked by reading, not run on hardware
//
#include <hip/hip_runtime.h>

typedef __attribute__((ext_vector_type(16))) __bf16 v16bf;
typedef __attribute__((ext_vector_type(8)))  float  v8f;

#define XP 264   // x LDS row pitch (256 + 8 pad), bf16 elems
#define WP1 264  // W1 stage pitch
#define WP2 72   // W2 stage pitch (64 + 8 pad)
#define HP 40    // h scratch pitch (32 + 8 pad)

#define SW1_BUF (64 * WP1)    // one W1 stage buffer, elems
#define SW2_BUF (256 * WP2)   // one W2 stage buffer, elems

union Frag16 { uint4 u[2]; v16bf v; };

__device__ __forceinline__ unsigned short f2bf(float f) {
  unsigned int u = __float_as_uint(f);
  u += 0x7FFFu + ((u >> 16) & 1u);   // round-to-nearest-even
  return (unsigned short)(u >> 16);
}

__device__ __forceinline__ float gelu_exact(float v) {
  return 0.5f * v * (1.0f + erff(v * 0.70710678118654752f));
}

// Async global -> LDS copy of 16 bytes per lane (GLOBAL_LOAD_ASYNC_TO_LDS_B128,
// tracked with ASYNCcnt; LDS byte address = low 32 bits of the flat pointer).
__device__ __forceinline__ void async_ld16(const unsigned short* g,
                                           unsigned short* l) {
  unsigned lds = (unsigned)(size_t)(void*)l;
  asm volatile("global_load_async_to_lds_b128 %0, %1, off"
               :: "v"(lds), "v"(g) : "memory");
}

// Load a 16x32 bf16 A/B fragment from LDS in the documented gfx1250 layout:
// lanes 0-15 hold K[0..7],K[16..23]; lanes 16-31 hold K[8..15],K[24..31];
// row index = lane % 16 (+ row0).
__device__ __forceinline__ v16bf ldsFrag(const unsigned short* base, int row0,
                                         int pitch, int k0) {
  int lane = threadIdx.x & 31;
  const unsigned short* p = base + (row0 + (lane & 15)) * pitch + k0 + ((lane >> 4) << 3);
  Frag16 f;
  f.u[0] = *(const uint4*)(p);
  f.u[1] = *(const uint4*)(p + 16);
  return f.v;
}

__device__ __forceinline__ v8f wmma_bf16(v16bf a, v16bf b, v8f c) {
  return __builtin_amdgcn_wmma_f32_16x16x32_bf16(false, a, false, b, (short)0, c,
                                                 false, false);
}

// ---------------------------------------------------------------------------
// Kernel 1: fp32 -> bf16 weight conversion (done once per launch into ws)
// ---------------------------------------------------------------------------
__global__ void convert_kernel(const float* __restrict__ src,
                               unsigned short* __restrict__ dst, int n) {
  int i = blockIdx.x * blockDim.x + threadIdx.x;
  int stride = gridDim.x * blockDim.x;
  for (int idx = i; idx < (n >> 2); idx += stride) {
    float4 a = ((const float4*)src)[idx];
    ushort4 b = make_ushort4(f2bf(a.x), f2bf(a.y), f2bf(a.z), f2bf(a.w));
    ((ushort4*)dst)[idx] = b;
  }
}

// ---------------------------------------------------------------------------
// Kernel 2: router.  One wave32 per token.
// ---------------------------------------------------------------------------
__global__ void router_kernel(const float* __restrict__ x,
                              const float* __restrict__ rw,   // (3,256)
                              const float* __restrict__ tg,   // (8,3)
                              float* __restrict__ gate,       // (N,8)
                              float* __restrict__ probsum,    // (8)
                              int N) {
  int wave = threadIdx.x >> 5;
  int lane = threadIdx.x & 31;
  int n = blockIdx.x * 8 + wave;

  __shared__ float ps[8];
  if (threadIdx.x < 8) ps[threadIdx.x] = 0.0f;
  __syncthreads();

  // z3 = x[n] . rw[k], lanes split the 256-dim dot (8 floats per lane)
  const float4* xv = (const float4*)(x + (size_t)n * 256);
  const float4* r0v = (const float4*)(rw);
  const float4* r1v = (const float4*)(rw + 256);
  const float4* r2v = (const float4*)(rw + 512);
  float z0 = 0.f, z1 = 0.f, z2 = 0.f;
#pragma unroll
  for (int j = 0; j < 2; ++j) {
    int c = lane * 2 + j;
    float4 a = xv[c];
    float4 b0 = r0v[c], b1 = r1v[c], b2 = r2v[c];
    z0 += a.x * b0.x + a.y * b0.y + a.z * b0.z + a.w * b0.w;
    z1 += a.x * b1.x + a.y * b1.y + a.z * b1.z + a.w * b1.w;
    z2 += a.x * b2.x + a.y * b2.y + a.z * b2.z + a.w * b2.w;
  }
#pragma unroll
  for (int m = 16; m >= 1; m >>= 1) {
    z0 += __shfl_xor(z0, m, 32);
    z1 += __shfl_xor(z1, m, 32);
    z2 += __shfl_xor(z2, m, 32);
  }

  if (lane == 0) {
    float p[8];
    float mx = -1e30f;
#pragma unroll
    for (int e = 0; e < 8; ++e) {
      float l = z0 * tg[e * 3] + z1 * tg[e * 3 + 1] + z2 * tg[e * 3 + 2];
      p[e] = l;
      mx = fmaxf(mx, l);
    }
    float s = 0.f;
#pragma unroll
    for (int e = 0; e < 8; ++e) { p[e] = __expf(p[e] - mx); s += p[e]; }
    float inv = 1.0f / s;
#pragma unroll
    for (int e = 0; e < 8; ++e) p[e] *= inv;
    // top-2 (first index wins on ties, matching jax top_k)
    int i1 = 0;
#pragma unroll
    for (int e = 1; e < 8; ++e) if (p[e] > p[i1]) i1 = e;
    int i2 = (i1 == 0) ? 1 : 0;
#pragma unroll
    for (int e = 0; e < 8; ++e) if (e != i1 && p[e] > p[i2]) i2 = e;
    float wsum = p[i1] + p[i2] + 1e-8f;
#pragma unroll
    for (int e = 0; e < 8; ++e) {
      float g = (e == i1) ? p[i1] / wsum : (e == i2) ? p[i2] / wsum : 0.0f;
      gate[(size_t)n * 8 + e] = g;
      atomicAdd(&ps[e], p[e]);
    }
  }
  __syncthreads();
  if (threadIdx.x < 8) atomicAdd(&probsum[threadIdx.x], ps[threadIdx.x]);
}

// ---------------------------------------------------------------------------
// Kernel 3: aux loss scalar
// ---------------------------------------------------------------------------
__global__ void aux_kernel(const float* __restrict__ probsum,
                           float* __restrict__ auxout, int N) {
  if (threadIdx.x == 0 && blockIdx.x == 0) {
    float acc = 0.f;
    for (int e = 0; e < 8; ++e) {
      float p = probsum[e] / (float)N;
      float d = p - 0.125f;
      acc += d * d;
    }
    *auxout = 0.01f * (acc * 0.125f);
  }
}

// ---------------------------------------------------------------------------
// Issue one double-buffered weight-chunk stage: 16 async b128 per wave
// (W1 rows [h0,h0+64) x 256, W2 rows [0,256) x cols [h0,h0+64)).
// ---------------------------------------------------------------------------
__device__ __forceinline__ void issue_stage(const unsigned short* __restrict__ W1b,
                                            const unsigned short* __restrict__ W2b,
                                            int e, int h0,
                                            unsigned short* sw1,
                                            unsigned short* sw2, int tid) {
  const unsigned short* W1e = W1b + (size_t)e * 262144;
  const unsigned short* W2e = W2b + (size_t)e * 262144;
#pragma unroll
  for (int j = 0; j < 8; ++j) {
    int idx = tid + j * 256;        // 2048 x b128 for W1 chunk
    int r = idx >> 5;               // 32 b128 per 256-elem row
    int c = (idx & 31) << 3;
    async_ld16(W1e + (size_t)(h0 + r) * 256 + c, sw1 + r * WP1 + c);
  }
#pragma unroll
  for (int j = 0; j < 8; ++j) {
    int idx = tid + j * 256;        // 2048 x b128 for W2 chunk
    int d = idx >> 3;               // 8 b128 per 64-elem row
    int c = (idx & 7) << 3;
    async_ld16(W2e + (size_t)d * 1024 + h0 + c, sw2 + d * WP2 + c);
  }
}

// ---------------------------------------------------------------------------
// Kernel 4: fused dense-gated MoE FFN.
// Block = 8 waves, each wave owns 16 tokens (128 tokens/block).
// grid.y splits H in halves; partial outputs combined with f32 atomics.
// Weights stream via double-buffered GLOBAL_LOAD_ASYNC_TO_LDS_B128.
// Buffer selection is pure pointer arithmetic off one LDS base so
// address-space inference keeps every fragment read on the DS path.
// ---------------------------------------------------------------------------
__global__ void __launch_bounds__(256) moe_ffn(
    const float* __restrict__ x,               // (N,256) fp32
    const unsigned short* __restrict__ W1b,    // (8,1024,256) bf16
    const unsigned short* __restrict__ W2b,    // (8,256,1024) bf16
    const float* __restrict__ gate,            // (N,8)
    float* __restrict__ out)                   // (N,256) f32 (pre-zeroed)
{
  extern __shared__ char smem[];
  unsigned short* sx   = (unsigned short*)smem;           // 128 x XP
  unsigned short* sw1base = sx + 128 * XP;                // 2 x (64 x WP1)
  unsigned short* sw2base = sw1base + 2 * SW1_BUF;        // 2 x (256 x WP2)
  unsigned short* sh      = sw2base + 2 * SW2_BUF;        // 8 x (16 x HP)

  const int tid = threadIdx.x;
  const int wave = tid >> 5;
  const int lane = tid & 31;
  const int tb = blockIdx.x * 128;           // block token base
  const int hbase = blockIdx.y * 512;        // this block's H half
  const int mb = (lane >> 4) << 3;           // C-frag row base: 0 or 8
  const int nloc = lane & 15;
  unsigned short* shw = sh + wave * 16 * HP;

  // prologue: kick off chunk 0 into buffer 0 before staging x
  issue_stage(W1b, W2b, 0, hbase, sw1base, sw2base, tid);

  // ---- stage x tile (fp32 -> bf16) into LDS, shared by all 8 waves
  for (int idx = tid; idx < 128 * 256 / 4; idx += 256) {
    int row = idx >> 6;
    int col = (idx & 63) << 2;
    float4 a = *(const float4*)(x + (size_t)(tb + row) * 256 + col);
    ushort4 b = make_ushort4(f2bf(a.x), f2bf(a.y), f2bf(a.z), f2bf(a.w));
    *(ushort4*)(sx + row * XP + col) = b;
  }

  v8f acc[16];
#pragma unroll
  for (int i = 0; i < 16; ++i) acc[i] = (v8f){0.f, 0.f, 0.f, 0.f, 0.f, 0.f, 0.f, 0.f};

  for (int e = 0; e < 8; ++e) {
    // per-token top-2 gates for this wave's 16 tokens (dense, 0 if unselected)
    float gv[8];
    float gs = 0.f;
#pragma unroll
    for (int i = 0; i < 8; ++i) {
      gv[i] = gate[(size_t)(tb + wave * 16 + mb + i) * 8 + e];
      gs += gv[i];
    }
    const bool active = (__ballot(gs > 0.0f) != 0ull);

    for (int hc = 0; hc < 8; ++hc) {
      const int ci = e * 8 + hc;
      const int p = ci & 1;
      unsigned short* sw1 = sw1base + p * SW1_BUF;   // GEP off LDS base
      unsigned short* sw2 = sw2base + p * SW2_BUF;

      __syncthreads();   // all waves done reading buffer p^1 (iteration ci-1)
      if (ci + 1 < 64) {
        const int nci = ci + 1;
        issue_stage(W1b, W2b, nci >> 3, hbase + (nci & 7) * 64,
                    sw1base + (p ^ 1) * SW1_BUF, sw2base + (p ^ 1) * SW2_BUF,
                    tid);
        // 16 loads of buffer p^1 in flight; wait for buffer p's 16 to land
        asm volatile("s_wait_asynccnt 0x10" ::: "memory");
      } else {
        asm volatile("s_wait_asynccnt 0x0" ::: "memory");
      }
      __syncthreads();   // buffer p visible to every wave

      if (active) {
#pragma unroll
        for (int hh = 0; hh < 2; ++hh) {   // two K32 groups inside the 64-chunk
          v8f hA = (v8f){0.f, 0.f, 0.f, 0.f, 0.f, 0.f, 0.f, 0.f};
          v8f hB = (v8f){0.f, 0.f, 0.f, 0.f, 0.f, 0.f, 0.f, 0.f};
          // GEMM1: h[16 x 32] = x[16 x 256] @ W1_chunk^T
#pragma unroll
          for (int kc = 0; kc < 8; ++kc) {
            v16bf A  = ldsFrag(sx, wave * 16, XP, kc * 32);
            v16bf B0 = ldsFrag(sw1, hh * 32, WP1, kc * 32);
            v16bf B1 = ldsFrag(sw1, hh * 32 + 16, WP1, kc * 32);
            hA = wmma_bf16(A, B0, hA);
            hB = wmma_bf16(A, B1, hB);
          }
          // gelu * gate, write to per-wave LDS scratch to convert
          // C-layout (lane=N, vgpr=M) into A-layout (lane=M, vgpr=K)
#pragma unroll
          for (int i = 0; i < 8; ++i) {
            int m = mb + i;
            float g = gv[i];
            shw[m * HP + nloc]      = f2bf(gelu_exact(hA[i]) * g);
            shw[m * HP + 16 + nloc] = f2bf(gelu_exact(hB[i]) * g);
          }
          asm volatile("s_wait_dscnt 0" ::: "memory");
          v16bf hF = ldsFrag(shw, 0, HP, 0);
          // GEMM2: out[16 x 256] += h[16 x 32] @ W2_chunk^T
#pragma unroll
          for (int dc = 0; dc < 16; ++dc) {
            v16bf B = ldsFrag(sw2, dc * 16, WP2, hh * 32);
            acc[dc] = wmma_bf16(hF, B, acc[dc]);
          }
        }
      }
    }
  }

  // accumulate partial (H-half) result into out
#pragma unroll
  for (int dc = 0; dc < 16; ++dc) {
#pragma unroll
    for (int i = 0; i < 8; ++i) {
      int row = tb + wave * 16 + mb + i;
      atomicAdd(&out[(size_t)row * 256 + dc * 16 + nloc], acc[dc][i]);
    }
  }
}

// ---------------------------------------------------------------------------
extern "C" void kernel_launch(void* const* d_in, const int* in_sizes, int n_in,
                              void* d_out, int out_size, void* d_ws, size_t ws_size,
                              hipStream_t stream) {
  const float* x  = (const float*)d_in[0];
  const float* W1 = (const float*)d_in[1];
  const float* W2 = (const float*)d_in[2];
  const float* rw = (const float*)d_in[3];
  const float* tg = (const float*)d_in[4];
  float* out = (float*)d_out;

  const int N = in_sizes[0] / 256;   // 16384 tokens
  const int W1N = in_sizes[1];       // 8*1024*256
  const int W2N = in_sizes[2];       // 8*256*1024

  unsigned short* W1b = (unsigned short*)d_ws;
  unsigned short* W2b = W1b + W1N;
  float* gate = (float*)(W2b + W2N);
  float* probsum = gate + (size_t)N * 8;

  // out = 0 (FFN accumulates atomically), probsum = 0
  hipMemsetAsync(d_out, 0, (size_t)out_size * sizeof(float), stream);
  hipMemsetAsync(probsum, 0, 8 * sizeof(float), stream);

  convert_kernel<<<2048, 256, 0, stream>>>(W1, W1b, W1N);
  convert_kernel<<<2048, 256, 0, stream>>>(W2, W2b, W2N);

  router_kernel<<<N / 8, 256, 0, stream>>>(x, rw, tg, gate, probsum, N);
  aux_kernel<<<1, 1, 0, stream>>>(probsum, out + (size_t)N * 256, N);

  size_t smem = (size_t)(128 * XP + 2 * SW1_BUF + 2 * SW2_BUF + 8 * 16 * HP) *
                sizeof(unsigned short);
  dim3 grid(N / 128, 2);
  moe_ffn<<<grid, 256, smem, stream>>>(x, W1b, W2b, gate, out);
}
